// SC_MIL_22308060135463
// MI455X (gfx1250) — compile-verified, matrix operand order checked by reading, not run on hardware
//
#include <hip/hip_runtime.h>
#include <hip/hip_bf16.h>
#include <stdint.h>

#define LL   16384
#define CC   1024
#define HH   8
#define DHH  128
#define NCC  64
#define BLKK 256

typedef __attribute__((ext_vector_type(16))) __bf16 bf16x16;
typedef __attribute__((ext_vector_type(8)))  float  f32x8;

union Frag {
  bf16x16 v;
  uint32_t u[8];
};

static __device__ __forceinline__ uint32_t bf16_1(float f) {
  uint32_t x = __builtin_bit_cast(uint32_t, f);
  return (x + 0x7FFFu + ((x >> 16) & 1u)) >> 16;
}
static __device__ __forceinline__ uint32_t bf16_pk(float lo, float hi) {
  return bf16_1(lo) | (bf16_1(hi) << 16);
}
static __device__ __forceinline__ f32x8 zero8() {
  f32x8 z = {0.f, 0.f, 0.f, 0.f, 0.f, 0.f, 0.f, 0.f};
  return z;
}
static __device__ __forceinline__ f32x8 wmma_bf16(const Frag& a, const Frag& b, f32x8 c) {
  return __builtin_amdgcn_wmma_f32_16x16x32_bf16(false, a.v, false, b.v, (short)0, c, false, false);
}
// k-pair VGPR index for 16-bit A/B fragments (ISA 7.12.2):
// lane<16: K pairs {0..7,16..23}/2 ; lane>=16: K pairs {8..15,24..31}/2
static __device__ __forceinline__ int kpv(int lane, int v) {
  int base = (lane >> 4) * 4;
  return (v < 4) ? (base + v) : (8 + base + (v - 4));
}

// Async global->LDS copies (ASYNCcnt path; LDS written directly, no VGPRs)
static __device__ __forceinline__ void async_copy_b32(uint32_t lds, const void* g) {
  asm volatile("global_load_async_to_lds_b32 %0, %1, off"
               :: "v"(lds), "v"((uint64_t)(uintptr_t)g) : "memory");
}
static __device__ __forceinline__ void async_copy_b128(uint32_t lds, const void* g) {
  asm volatile("global_load_async_to_lds_b128 %0, %1, off"
               :: "v"(lds), "v"((uint64_t)(uintptr_t)g) : "memory");
}
static __device__ __forceinline__ void wait_async0() {
  asm volatile("s_wait_asynccnt 0" ::: "memory");
}

// ---------------------------------------------------------------------------
// Stable counting-sort permutation: order[c*256 + j] = j-th index with label c
// ---------------------------------------------------------------------------
__global__ void k_order(const int* __restrict__ labels, int* __restrict__ order) {
  int c = threadIdx.x;
  if (c >= NCC) return;
  int w = c * BLKK;
  for (int i = 0; i < LL; ++i)
    if (labels[i] == c) order[w++] = i;
}

// ---------------------------------------------------------------------------
// GEMM 1: h = relu(x @ W_fc1 + b).  f32 in, bf16 WMMA, f32 out.
// Block tile 128x128, BK=32, 8 waves (2x4), wave tile 64x32.
// ---------------------------------------------------------------------------
__global__ __launch_bounds__(256) void k_gemm_fc1(
    const float* __restrict__ A, const float* __restrict__ W,
    const float* __restrict__ bias, float* __restrict__ out,
    int M, int N, int K) {
  __shared__ uint32_t As[128 * 16];
  __shared__ uint32_t Bs[16 * 128];
  const int tid = threadIdx.x, lane = tid & 31, wid = tid >> 5;
  const int wm = wid >> 2, wn = wid & 3;
  const int rowBase = blockIdx.y * 128, colBase = blockIdx.x * 128;
  f32x8 acc[4][2];
#pragma unroll
  for (int i = 0; i < 4; ++i)
#pragma unroll
    for (int j = 0; j < 2; ++j) acc[i][j] = zero8();

  for (int ks = 0; ks < K; ks += 32) {
    for (int i = tid; i < 2048; i += 256) {
      int r = i >> 4, kp = i & 15;
      const float* p = A + (size_t)(rowBase + r) * K + ks + kp * 2;
      As[i] = bf16_pk(p[0], p[1]);
    }
    for (int i = tid; i < 2048; i += 256) {
      int kp = i >> 7, c = i & 127;
      const float* p = W + (size_t)(ks + kp * 2) * N + colBase + c;
      Bs[i] = bf16_pk(p[0], p[N]);
    }
    __syncthreads();
    Frag a[4], b[2];
#pragma unroll
    for (int mt = 0; mt < 4; ++mt) {
      int r = wm * 64 + mt * 16 + (lane & 15);
#pragma unroll
      for (int v = 0; v < 8; ++v) a[mt].u[v] = As[r * 16 + kpv(lane, v)];
    }
#pragma unroll
    for (int nt = 0; nt < 2; ++nt) {
      int c = wn * 32 + nt * 16 + (lane & 15);
#pragma unroll
      for (int v = 0; v < 8; ++v) b[nt].u[v] = Bs[kpv(lane, v) * 128 + c];
    }
#pragma unroll
    for (int mt = 0; mt < 4; ++mt)
#pragma unroll
      for (int nt = 0; nt < 2; ++nt) acc[mt][nt] = wmma_bf16(a[mt], b[nt], acc[mt][nt]);
    __syncthreads();
  }
#pragma unroll
  for (int mt = 0; mt < 4; ++mt)
#pragma unroll
    for (int nt = 0; nt < 2; ++nt)
#pragma unroll
      for (int v = 0; v < 8; ++v) {
        int gr = rowBase + wm * 64 + mt * 16 + v + 8 * (lane >> 4);
        int gc = colBase + wn * 32 + nt * 16 + (lane & 15);
        float val = acc[mt][nt][v] + bias[gc];
        out[(size_t)gr * N + gc] = fmaxf(val, 0.f);
      }
}

// ---------------------------------------------------------------------------
// Row gather: xs[i] = h[order[i]]
// ---------------------------------------------------------------------------
__global__ void k_gather(const float4* __restrict__ h4, const int* __restrict__ order,
                         float4* __restrict__ xs4) {
  int i = blockIdx.x;
  int r = order[i];
  xs4[(size_t)i * 256 + threadIdx.x] = h4[(size_t)r * 256 + threadIdx.x];
}

// ---------------------------------------------------------------------------
// GEMM 2: qkv(bf16) = xs @ W_qkv + b  (N = 3072)
// ---------------------------------------------------------------------------
__global__ __launch_bounds__(256) void k_gemm_qkv(
    const float* __restrict__ A, const float* __restrict__ W,
    const float* __restrict__ bias, uint16_t* __restrict__ out,
    int M, int N, int K) {
  __shared__ uint32_t As[128 * 16];
  __shared__ uint32_t Bs[16 * 128];
  const int tid = threadIdx.x, lane = tid & 31, wid = tid >> 5;
  const int wm = wid >> 2, wn = wid & 3;
  const int rowBase = blockIdx.y * 128, colBase = blockIdx.x * 128;
  f32x8 acc[4][2];
#pragma unroll
  for (int i = 0; i < 4; ++i)
#pragma unroll
    for (int j = 0; j < 2; ++j) acc[i][j] = zero8();

  for (int ks = 0; ks < K; ks += 32) {
    for (int i = tid; i < 2048; i += 256) {
      int r = i >> 4, kp = i & 15;
      const float* p = A + (size_t)(rowBase + r) * K + ks + kp * 2;
      As[i] = bf16_pk(p[0], p[1]);
    }
    for (int i = tid; i < 2048; i += 256) {
      int kp = i >> 7, c = i & 127;
      const float* p = W + (size_t)(ks + kp * 2) * N + colBase + c;
      Bs[i] = bf16_pk(p[0], p[N]);
    }
    __syncthreads();
    Frag a[4], b[2];
#pragma unroll
    for (int mt = 0; mt < 4; ++mt) {
      int r = wm * 64 + mt * 16 + (lane & 15);
#pragma unroll
      for (int v = 0; v < 8; ++v) a[mt].u[v] = As[r * 16 + kpv(lane, v)];
    }
#pragma unroll
    for (int nt = 0; nt < 2; ++nt) {
      int c = wn * 32 + nt * 16 + (lane & 15);
#pragma unroll
      for (int v = 0; v < 8; ++v) b[nt].u[v] = Bs[kpv(lane, v) * 128 + c];
    }
#pragma unroll
    for (int mt = 0; mt < 4; ++mt)
#pragma unroll
      for (int nt = 0; nt < 2; ++nt) acc[mt][nt] = wmma_bf16(a[mt], b[nt], acc[mt][nt]);
    __syncthreads();
  }
#pragma unroll
  for (int mt = 0; mt < 4; ++mt)
#pragma unroll
    for (int nt = 0; nt < 2; ++nt)
#pragma unroll
      for (int v = 0; v < 8; ++v) {
        int gr = rowBase + wm * 64 + mt * 16 + v + 8 * (lane >> 4);
        int gc = colBase + wn * 32 + nt * 16 + (lane & 15);
        out[(size_t)gr * N + gc] = (uint16_t)bf16_1(acc[mt][nt][v] + bias[gc]);
      }
}

// ---------------------------------------------------------------------------
// Block-local attention (flash-style, online softmax over 2 column halves).
// Grid = H * NC * 2 (row-halves). 8 waves; wave owns 16 query rows.
// qkv is bf16 [L][3072]; output attn bf16 [L][1024] (col = h*128+d).
// K^T tiles are staged with GLOBAL_LOAD_ASYNC_TO_LDS (pure u32 copies of the
// pre-packed bf16 d-pairs); V tiles need row-pair packing so stay on VALU.
// ---------------------------------------------------------------------------
__global__ __launch_bounds__(256) void k_attn(const uint16_t* __restrict__ qkv16,
                                              uint16_t* __restrict__ attn16) {
  __shared__ uint32_t KtP[64 * 128];  // K^T d-pairs, reused as P (bf16) buffer
  __shared__ uint32_t Vp[64 * 128];   // V row-pairs
  const int tid = threadIdx.x, lane = tid & 31, wid = tid >> 5;
  const int bid = blockIdx.x;
  const int h = bid >> 7;
  const int rem = bid & 127;
  const int chunk = rem >> 1;
  const int rhalf = rem & 1;
  const int l0 = chunk * BLKK;
  const int qrowBase = l0 + rhalf * 128;
  const uint32_t* qkv32 = (const uint32_t*)qkv16;

  // Preload Q fragments for all 4 k-steps (d = 0..127), reused for both halves.
  Frag qf[4];
  {
    int r = qrowBase + wid * 16 + (lane & 15);
    size_t base = (size_t)r * 1536 + (size_t)h * 64;  // u32 units
#pragma unroll
    for (int ks4 = 0; ks4 < 4; ++ks4)
#pragma unroll
      for (int v = 0; v < 8; ++v)
        qf[ks4].u[v] = qkv32[base + ks4 * 16 + kpv(lane, v)];
  }

  f32x8 Oacc[8];
#pragma unroll
  for (int i = 0; i < 8; ++i) Oacc[i] = zero8();
  float mrow[8], lrow[8];
#pragma unroll
  for (int v = 0; v < 8; ++v) { mrow[v] = -1e30f; lrow[v] = 0.f; }
  const float scale = 0.088388347648318447f;  // 1/sqrt(128)

  const uint32_t ldsKt = (uint32_t)(uintptr_t)(&KtP[0]);

  for (int halfI = 0; halfI < 2; ++halfI) {
    __syncthreads();  // previous half's P/V reads complete before restaging
    const int c0 = halfI * 128;
    // Stage K^T as packed d-pairs via async copy: KtP[dp*128 + c]
    {
      const uint32_t* gbase =
          qkv32 + (size_t)(l0 + c0) * 1536 + 512 + (size_t)h * 64;
      for (int i = tid; i < 8192; i += 256) {
        int c = i >> 6, dp = i & 63;
        async_copy_b32(ldsKt + (uint32_t)(dp * 128 + c) * 4u,
                       gbase + (size_t)c * 1536 + dp);
      }
    }
    // Stage V as packed row-pairs: Vp[cp*128 + d] = {V[2cp][d], V[2cp+1][d]}
    for (int i = tid; i < 8192; i += 256) {
      int cp = i >> 7, d = i & 127;
      const uint16_t* p =
          qkv16 + (size_t)(l0 + c0 + 2 * cp) * 3072 + 2048 + h * 128 + d;
      Vp[i] = (uint32_t)p[0] | ((uint32_t)p[3072] << 16);
    }
    wait_async0();
    __syncthreads();

    // S = Q K^T for this wave's 16 rows x 128 cols
    f32x8 S[8];
#pragma unroll
    for (int i = 0; i < 8; ++i) S[i] = zero8();
#pragma unroll
    for (int ks4 = 0; ks4 < 4; ++ks4) {
#pragma unroll
      for (int nt = 0; nt < 8; ++nt) {
        Frag b;
        int cc = nt * 16 + (lane & 15);
#pragma unroll
        for (int v = 0; v < 8; ++v)
          b.u[v] = KtP[(ks4 * 16 + kpv(lane, v)) * 128 + cc];
        S[nt] = wmma_bf16(qf[ks4], b, S[nt]);
      }
    }
    // Online softmax update. Row of element (nt, v) = v + 8*(lane>>4); its
    // 16 columns live in lanes sharing (lane>>4) -> reduce with xor masks 1..8.
    float mloc[8], rsum[8], mnew[8], corr[8];
#pragma unroll
    for (int v = 0; v < 8; ++v) mloc[v] = -1e30f;
#pragma unroll
    for (int nt = 0; nt < 8; ++nt)
#pragma unroll
      for (int v = 0; v < 8; ++v) {
        S[nt][v] *= scale;
        mloc[v] = fmaxf(mloc[v], S[nt][v]);
      }
#pragma unroll
    for (int v = 0; v < 8; ++v) {
#pragma unroll
      for (int m = 1; m < 16; m <<= 1)
        mloc[v] = fmaxf(mloc[v], __shfl_xor(mloc[v], m, 32));
      mnew[v] = fmaxf(mrow[v], mloc[v]);
      corr[v] = expf(mrow[v] - mnew[v]);
      rsum[v] = 0.f;
    }
#pragma unroll
    for (int nt = 0; nt < 8; ++nt)
#pragma unroll
      for (int v = 0; v < 8; ++v) {
        float p = expf(S[nt][v] - mnew[v]);
        S[nt][v] = p;
        rsum[v] += p;
      }
#pragma unroll
    for (int v = 0; v < 8; ++v) {
#pragma unroll
      for (int m = 1; m < 16; m <<= 1) rsum[v] += __shfl_xor(rsum[v], m, 32);
      lrow[v] = lrow[v] * corr[v] + rsum[v];
      mrow[v] = mnew[v];
    }
#pragma unroll
    for (int nt = 0; nt < 8; ++nt)
#pragma unroll
      for (int v = 0; v < 8; ++v) Oacc[nt][v] *= corr[v];

    __syncthreads();  // all waves finished reading KtP before P overwrite
    // Write P (bf16) into per-wave region of KtP: [16 rows][128 cols]
    {
      uint16_t* P16 = (uint16_t*)KtP + wid * 2048;
#pragma unroll
      for (int nt = 0; nt < 8; ++nt)
#pragma unroll
        for (int v = 0; v < 8; ++v) {
          int row = v + 8 * (lane >> 4);
          int col = nt * 16 + (lane & 15);
          P16[row * 128 + col] = (uint16_t)bf16_1(S[nt][v]);
        }
    }
    // O += P x V (wave reads only its own P region; LDS is in-order per wave)
    const uint32_t* Pw = KtP + wid * 1024;
#pragma unroll
    for (int kc4 = 0; kc4 < 4; ++kc4) {
      Frag a;
      int row = lane & 15;
#pragma unroll
      for (int v = 0; v < 8; ++v)
        a.u[v] = Pw[row * 64 + kc4 * 16 + kpv(lane, v)];
#pragma unroll
      for (int nt = 0; nt < 8; ++nt) {
        Frag b;
        int d = nt * 16 + (lane & 15);
#pragma unroll
        for (int v = 0; v < 8; ++v)
          b.u[v] = Vp[(kc4 * 16 + kpv(lane, v)) * 128 + d];
        Oacc[nt] = wmma_bf16(a, b, Oacc[nt]);
      }
    }
  }
  // Normalize and store bf16 into attn[L][1024] at col = h*128 + d
#pragma unroll
  for (int v = 0; v < 8; ++v) {
    float inv = 1.f / lrow[v];
#pragma unroll
    for (int nt = 0; nt < 8; ++nt) {
      int gr = qrowBase + wid * 16 + v + 8 * (lane >> 4);
      int gc = h * 128 + nt * 16 + (lane & 15);
      attn16[(size_t)gr * 1024 + gc] = (uint16_t)bf16_1(Oacc[nt][v] * inv);
    }
  }
}

// ---------------------------------------------------------------------------
// GEMM 3: hs = attn(bf16) @ W_out + b_out + xs   (f32 out)
// A tiles are already bf16 -> staged via async B128 copies (ASYNCcnt path).
// ---------------------------------------------------------------------------
__global__ __launch_bounds__(256) void k_gemm_out(
    const uint32_t* __restrict__ A32, const float* __restrict__ W,
    const float* __restrict__ bias, const float* __restrict__ resid,
    float* __restrict__ out, int M, int N, int K) {
  __shared__ uint32_t As[128 * 16];
  __shared__ uint32_t Bs[16 * 128];
  const int tid = threadIdx.x, lane = tid & 31, wid = tid >> 5;
  const int wm = wid >> 2, wn = wid & 3;
  const int rowBase = blockIdx.y * 128, colBase = blockIdx.x * 128;
  const uint32_t ldsAs = (uint32_t)(uintptr_t)(&As[0]);
  f32x8 acc[4][2];
#pragma unroll
  for (int i = 0; i < 4; ++i)
#pragma unroll
    for (int j = 0; j < 2; ++j) acc[i][j] = zero8();

  for (int ks = 0; ks < K; ks += 32) {
    // A: 128 rows x 16 u32, each row = 4 x 16B async chunks
    for (int i = tid; i < 512; i += 256) {
      int r = i >> 2, q = i & 3;
      async_copy_b128(ldsAs + (uint32_t)(r * 16 + q * 4) * 4u,
                      A32 + (size_t)(rowBase + r) * (K / 2) + (ks >> 1) + q * 4);
    }
    for (int i = tid; i < 2048; i += 256) {
      int kp = i >> 7, c = i & 127;
      const float* p = W + (size_t)(ks + kp * 2) * N + colBase + c;
      Bs[i] = bf16_pk(p[0], p[N]);
    }
    wait_async0();
    __syncthreads();
    Frag a[4], b[2];
#pragma unroll
    for (int mt = 0; mt < 4; ++mt) {
      int r = wm * 64 + mt * 16 + (lane & 15);
#pragma unroll
      for (int v = 0; v < 8; ++v) a[mt].u[v] = As[r * 16 + kpv(lane, v)];
    }
#pragma unroll
    for (int nt = 0; nt < 2; ++nt) {
      int c = wn * 32 + nt * 16 + (lane & 15);
#pragma unroll
      for (int v = 0; v < 8; ++v) b[nt].u[v] = Bs[kpv(lane, v) * 128 + c];
    }
#pragma unroll
    for (int mt = 0; mt < 4; ++mt)
#pragma unroll
      for (int nt = 0; nt < 2; ++nt) acc[mt][nt] = wmma_bf16(a[mt], b[nt], acc[mt][nt]);
    __syncthreads();
  }
#pragma unroll
  for (int mt = 0; mt < 4; ++mt)
#pragma unroll
    for (int nt = 0; nt < 2; ++nt)
#pragma unroll
      for (int v = 0; v < 8; ++v) {
        int gr = rowBase + wm * 64 + mt * 16 + v + 8 * (lane >> 4);
        int gc = colBase + wn * 32 + nt * 16 + (lane & 15);
        out[(size_t)gr * N + gc] =
            acc[mt][nt][v] + bias[gc] + resid[(size_t)gr * N + gc];
      }
}

// ---------------------------------------------------------------------------
// GEMM 4 (fused gate): pre = hs @ [W_aa | W_ab] (N=512);
// A_logits[row] = sum_c tanh(pre_a)*sigmoid(pre_g) * W_ac[c] + b_ac
// Wave owns 16 rows x all 512 cols; col c and c+256 share lane & slot.
// ---------------------------------------------------------------------------
__global__ __launch_bounds__(256) void k_gemm_ag(
    const float* __restrict__ hs, const float* __restrict__ Waa,
    const float* __restrict__ Wab, const float* __restrict__ baa,
    const float* __restrict__ bab, const float* __restrict__ Wac,
    const float* __restrict__ bac, float* __restrict__ alog) {
  __shared__ uint32_t As[128 * 16];  // 8 KB
  __shared__ uint32_t Bs[16 * 512];  // 32 KB
  const int K = 1024;
  const int tid = threadIdx.x, lane = tid & 31, wid = tid >> 5;
  const int rowBase = blockIdx.x * 128;
  f32x8 acc[32];
#pragma unroll
  for (int i = 0; i < 32; ++i) acc[i] = zero8();

  for (int ks = 0; ks < K; ks += 32) {
    for (int i = tid; i < 2048; i += 256) {
      int r = i >> 4, kp = i & 15;
      const float* p = hs + (size_t)(rowBase + r) * K + ks + kp * 2;
      As[i] = bf16_pk(p[0], p[1]);
    }
    for (int i = tid; i < 8192; i += 256) {
      int kp = i >> 9, c = i & 511;
      const float* p = (c < 256) ? (Waa + (size_t)(ks + kp * 2) * 256 + c)
                                 : (Wab + (size_t)(ks + kp * 2) * 256 + (c - 256));
      Bs[i] = bf16_pk(p[0], p[256]);
    }
    __syncthreads();
    Frag a;
    {
      int r = wid * 16 + (lane & 15);
#pragma unroll
      for (int v = 0; v < 8; ++v) a.u[v] = As[r * 16 + kpv(lane, v)];
    }
#pragma unroll
    for (int t = 0; t < 32; ++t) {
      Frag b;
      int c = t * 16 + (lane & 15);
#pragma unroll
      for (int v = 0; v < 8; ++v) b.u[v] = Bs[kpv(lane, v) * 512 + c];
      acc[t] = wmma_bf16(a, b, acc[t]);
    }
    __syncthreads();
  }
  float part[8] = {0.f, 0.f, 0.f, 0.f, 0.f, 0.f, 0.f, 0.f};
  const int c16 = lane & 15;
#pragma unroll
  for (int t = 0; t < 16; ++t) {
    int c = t * 16 + c16;
    float wac = Wac[c], ba = baa[c], bg = bab[c];
#pragma unroll
    for (int v = 0; v < 8; ++v) {
      float pa = acc[t][v] + ba;
      float pg = acc[t + 16][v] + bg;
      float s = tanhf(pa) * (1.f / (1.f + expf(-pg)));
      part[v] += s * wac;
    }
  }
#pragma unroll
  for (int v = 0; v < 8; ++v) {
#pragma unroll
    for (int m = 1; m < 16; m <<= 1) part[v] += __shfl_xor(part[v], m, 32);
    if (c16 == 0) {
      int gr = rowBase + wid * 16 + v + 8 * (lane >> 4);
      alog[gr] = part[v] + bac[0];
    }
  }
}

// ---------------------------------------------------------------------------
// Softmax stats over L, softmax-weighted bag, final classifier
// ---------------------------------------------------------------------------
__global__ __launch_bounds__(256) void k_softmax_stats(const float* __restrict__ alog,
                                                       float* __restrict__ stats) {
  __shared__ float red[256];
  const int tid = threadIdx.x;
  float m = -1e30f;
  for (int i = tid; i < LL; i += 256) m = fmaxf(m, alog[i]);
  red[tid] = m;
  __syncthreads();
  for (int s = 128; s > 0; s >>= 1) {
    if (tid < s) red[tid] = fmaxf(red[tid], red[tid + s]);
    __syncthreads();
  }
  float mx = red[0];
  __syncthreads();
  float z = 0.f;
  for (int i = tid; i < LL; i += 256) z += expf(alog[i] - mx);
  red[tid] = z;
  __syncthreads();
  for (int s = 128; s > 0; s >>= 1) {
    if (tid < s) red[tid] += red[tid + s];
    __syncthreads();
  }
  if (tid == 0) {
    stats[0] = mx;
    stats[1] = red[0];
  }
}

__global__ __launch_bounds__(256) void k_bag(const float* __restrict__ alog,
                                             const float* __restrict__ stats,
                                             const float* __restrict__ hs,
                                             float* __restrict__ bag) {
  __shared__ float w[256];
  const int col = blockIdx.x * 256 + threadIdx.x;
  const float m = stats[0];
  const float invZ = 1.f / stats[1];
  float acc = 0.f;
  for (int base = 0; base < LL; base += 256) {
    __syncthreads();
    w[threadIdx.x] = expf(alog[base + threadIdx.x] - m);
    __syncthreads();
    for (int j = 0; j < 256; ++j)
      acc += w[j] * hs[(size_t)(base + j) * CC + col];
  }
  bag[col] = acc * invZ;
}

__global__ __launch_bounds__(256) void k_cls(const float* __restrict__ bag,
                                             const float* __restrict__ Wcls,
                                             const float* __restrict__ bcls,
                                             float* __restrict__ out) {
  __shared__ float r0[256], r1[256];
  const int tid = threadIdx.x;
  float a0 = 0.f, a1 = 0.f;
  for (int c = tid; c < CC; c += 256) {
    float b = bag[c];
    a0 += b * Wcls[c * 2 + 0];
    a1 += b * Wcls[c * 2 + 1];
  }
  r0[tid] = a0;
  r1[tid] = a1;
  __syncthreads();
  for (int s = 128; s > 0; s >>= 1) {
    if (tid < s) {
      r0[tid] += r0[tid + s];
      r1[tid] += r1[tid + s];
    }
    __syncthreads();
  }
  if (tid == 0) {
    out[0] = r0[0] + bcls[0];
    out[1] = r1[0] + bcls[1];
  }
}

// ---------------------------------------------------------------------------
extern "C" void kernel_launch(void* const* d_in, const int* in_sizes, int n_in,
                              void* d_out, int out_size, void* d_ws, size_t ws_size,
                              hipStream_t stream) {
  const float* x      = (const float*)d_in[0];
  const int*   labels = (const int*)d_in[1];
  const float* W_fc1  = (const float*)d_in[2];
  const float* b_fc1  = (const float*)d_in[3];
  const float* W_qkv  = (const float*)d_in[4];
  const float* b_qkv  = (const float*)d_in[5];
  const float* W_out  = (const float*)d_in[6];
  const float* b_out  = (const float*)d_in[7];
  const float* W_aa   = (const float*)d_in[8];
  const float* b_aa   = (const float*)d_in[9];
  const float* W_ab   = (const float*)d_in[10];
  const float* b_ab   = (const float*)d_in[11];
  const float* W_ac   = (const float*)d_in[12];
  const float* b_ac   = (const float*)d_in[13];
  const float* W_cls  = (const float*)d_in[14];
  const float* b_cls  = (const float*)d_in[15];

  char* ws = (char*)d_ws;
  const size_t OFF_H   = 0;                         // h, later reused as hs (64 MB)
  const size_t OFF_XS  = OFF_H + 67108864;          // xs f32 (64 MB)
  const size_t OFF_QKV = OFF_XS + 67108864;         // qkv bf16 (96 MB)
  const size_t OFF_ATT = OFF_QKV + 100663296;       // attn bf16 (32 MB)
  const size_t OFF_ORD = OFF_ATT + 33554432;        // order i32 (64 KB)
  const size_t OFF_AL  = OFF_ORD + 65536;           // A_logits f32 (64 KB)
  const size_t OFF_ST  = OFF_AL + 65536;            // softmax stats
  const size_t OFF_BAG = OFF_ST + 256;              // bag f32 (4 KB)

  float*    h_hs   = (float*)(ws + OFF_H);
  float*    xs     = (float*)(ws + OFF_XS);
  uint16_t* qkv16  = (uint16_t*)(ws + OFF_QKV);
  uint16_t* attn16 = (uint16_t*)(ws + OFF_ATT);
  int*      order  = (int*)(ws + OFF_ORD);
  float*    alog   = (float*)(ws + OFF_AL);
  float*    stats  = (float*)(ws + OFF_ST);
  float*    bag    = (float*)(ws + OFF_BAG);

  k_order<<<1, 64, 0, stream>>>(labels, order);
  k_gemm_fc1<<<dim3(CC / 128, LL / 128), 256, 0, stream>>>(x, W_fc1, b_fc1, h_hs,
                                                           LL, CC, CC);
  k_gather<<<LL, 256, 0, stream>>>((const float4*)h_hs, order, (float4*)xs);
  k_gemm_qkv<<<dim3(3 * CC / 128, LL / 128), 256, 0, stream>>>(xs, W_qkv, b_qkv,
                                                               qkv16, LL, 3 * CC, CC);
  k_attn<<<HH * NCC * 2, 256, 0, stream>>>(qkv16, attn16);
  k_gemm_out<<<dim3(CC / 128, LL / 128), 256, 0, stream>>>(
      (const uint32_t*)attn16, W_out, b_out, xs, h_hs, LL, CC, CC);
  k_gemm_ag<<<LL / 128, 256, 0, stream>>>(h_hs, W_aa, W_ab, b_aa, b_ab, W_ac, b_ac,
                                          alog);
  k_softmax_stats<<<1, 256, 0, stream>>>(alog, stats);
  k_bag<<<CC / 256, 256, 0, stream>>>(alog, stats, h_hs, bag);
  k_cls<<<1, 256, 0, stream>>>(bag, W_cls, b_cls, (float*)d_out);
}
